// SlotPath_44032004718750
// MI455X (gfx1250) — compile-verified, matrix operand order checked by reading, not run on hardware
//
#include <hip/hip_runtime.h>
#include <hip/hip_bf16.h>
#include <math.h>

// ---------------------------------------------------------------------------
// Problem constants (match reference)
// ---------------------------------------------------------------------------
#define Bsz   2048
#define NS    128
#define Dd    1024
#define TOPK  8
#define DH    512
#define D2    2048        // 2*D
#define D3    3072        // 3*D
#define BK    (Bsz*TOPK)  // 16384 gathered rows

typedef __attribute__((ext_vector_type(16))) __bf16 v16bf;
typedef __attribute__((ext_vector_type(8)))  __bf16 v8bf;
typedef __attribute__((ext_vector_type(8)))  float  v8f;

#define WMMA_BF16(a, b, c) \
  __builtin_amdgcn_wmma_f32_16x16x32_bf16(false, (a), false, (b), (short)0, (c), false, false)

// ---------------------------------------------------------------------------
// Fragment loader: row-major [rows, ld] bf16 matrix, 16x32 tile at
// (rowbase, k0).  Per ISA 16-bit A/B layout: lanes 0-15 hold row (rowbase+lane),
// K = k0..k0+7 (VGPR0-3) and k0+16..k0+23 (VGPR4-7); lanes 16-31 hold
// K = k0+8.. and k0+24.. .  Two 16-byte loads per lane.
// ---------------------------------------------------------------------------
__device__ __forceinline__ v16bf load_frag(const __bf16* __restrict__ mat,
                                           int rowbase, int ld, int k0, int lane) {
  const __bf16* p = mat + (size_t)(rowbase + (lane & 15)) * ld + k0 + ((lane >> 4) << 3);
  v8bf lo = *(const v8bf*)p;
  v8bf hi = *(const v8bf*)(p + 16);
  return __builtin_shufflevector(lo, hi, 0,1,2,3,4,5,6,7,8,9,10,11,12,13,14,15);
}

// ---------------------------------------------------------------------------
// Generic WMMA GEMM:  C[M,N] = epi( A[M,K](bf16, lda) @ B[N,K]^T(bf16, ldb) + bias )
// Wave tile: 16 rows x 64 cols (4 accumulators, A fragment reused x4).
// Block = 8 waves stacked in M -> 128 rows x 64 cols per block.
// EPI: 0 = f32 store (optional *1/(|tau|+0.1)),  1 = bf16 store,  2 = exact-GELU -> bf16
// ---------------------------------------------------------------------------
template <int EPI>
__global__ __launch_bounds__(256) void wmma_gemm_kernel(
    const __bf16* __restrict__ A, int lda,
    const __bf16* __restrict__ B, int ldb,
    const float* __restrict__ bias,
    const float* __restrict__ tau,
    float* __restrict__ Cf, __bf16* __restrict__ Cb, int ldc, int K) {
  const int lane = threadIdx.x & 31;
  const int wave = threadIdx.x >> 5;
  const int m0 = (blockIdx.y * 8 + wave) * 16;
  const int n0 = blockIdx.x * 64;

  v8f acc0 = {}, acc1 = {}, acc2 = {}, acc3 = {};
  for (int k0 = 0; k0 < K; k0 += 32) {
    v16bf a  = load_frag(A, m0,       lda, k0, lane);
    v16bf b0 = load_frag(B, n0,       ldb, k0, lane);
    v16bf b1 = load_frag(B, n0 + 16,  ldb, k0, lane);
    v16bf b2 = load_frag(B, n0 + 32,  ldb, k0, lane);
    v16bf b3 = load_frag(B, n0 + 48,  ldb, k0, lane);
    acc0 = WMMA_BF16(a, b0, acc0);
    acc1 = WMMA_BF16(a, b1, acc1);
    acc2 = WMMA_BF16(a, b2, acc2);
    acc3 = WMMA_BF16(a, b3, acc3);
  }

  float scale = 1.f;
  if (EPI == 0 && tau) scale = 1.f / (fabsf(*tau) + 0.1f);

  const int hi = lane >> 4, nn = lane & 15;
  v8f accs[4] = {acc0, acc1, acc2, acc3};
#pragma unroll
  for (int c = 0; c < 4; ++c) {
    const int n = n0 + c * 16 + nn;
    const float bv = bias ? bias[n] : 0.f;
#pragma unroll
    for (int r = 0; r < 8; ++r) {
      const int m = m0 + r + (hi << 3);
      float v = accs[c][r] + bv;
      if (EPI == 0) {
        Cf[(size_t)m * ldc + n] = v * scale;
      } else if (EPI == 1) {
        Cb[(size_t)m * ldc + n] = (__bf16)v;
      } else {  // exact GELU: x * 0.5 * (1 + erf(x/sqrt(2)))
        v = 0.5f * v * (1.f + erff(v * 0.70710678118654752f));
        Cb[(size_t)m * ldc + n] = (__bf16)v;
      }
    }
  }
}

// ---------------------------------------------------------------------------
// Fused 3-gate GRU GEMM + epilogue.
// Rows m = b*8+k over gathered states (A = ss_bf [16384,1024]).
// Wave tile: 16 rows x 32 d-cols, 6 accumulators (r/z/n gates x 2 col tiles)
// reading whh rows {d, d+16} + {1024,2048} offsets.  A fragment reused x6.
// Epilogue: GRU gates, scatter updated f32 into S_new, per-lane weighted sum
// into U (each (b,d) owned by exactly one lane of one wave -> no atomics).
// ---------------------------------------------------------------------------
__global__ __launch_bounds__(256) void gru_fused_kernel(
    const __bf16* __restrict__ ssb,   // [16384,1024]
    const __bf16* __restrict__ whhb,  // [3072,1024]
    const float* __restrict__ bhh,    // [3072]
    const float* __restrict__ gi,     // [2048,3072] (= x@wih^T + bih)
    const float* __restrict__ S,      // [2048,128,1024]
    const int* __restrict__ tidx,     // [16384]
    const float* __restrict__ tw,     // [16384] softmax weights
    float* __restrict__ Snew,         // [2048,128,1024]
    __bf16* __restrict__ Ub) {        // [2048,1024]
  const int lane = threadIdx.x & 31;
  const int wave = threadIdx.x >> 5;
  const int m0 = (blockIdx.y * 8 + wave) * 16;  // gridDim.y = 128
  const int d0 = blockIdx.x * 32;               // gridDim.x = 32

  v8f ar0 = {}, az0 = {}, an0 = {};
  v8f ar1 = {}, az1 = {}, an1 = {};
  for (int k0 = 0; k0 < Dd; k0 += 32) {
    v16bf a   = load_frag(ssb,  m0,             Dd, k0, lane);
    v16bf br0 = load_frag(whhb, d0,             Dd, k0, lane);
    v16bf br1 = load_frag(whhb, d0 + 16,        Dd, k0, lane);
    v16bf bz0 = load_frag(whhb, d0 + Dd,        Dd, k0, lane);
    v16bf bz1 = load_frag(whhb, d0 + Dd + 16,   Dd, k0, lane);
    v16bf bn0 = load_frag(whhb, d0 + 2*Dd,      Dd, k0, lane);
    v16bf bn1 = load_frag(whhb, d0 + 2*Dd + 16, Dd, k0, lane);
    ar0 = WMMA_BF16(a, br0, ar0);
    ar1 = WMMA_BF16(a, br1, ar1);
    az0 = WMMA_BF16(a, bz0, az0);
    az1 = WMMA_BF16(a, bz1, az1);
    an0 = WMMA_BF16(a, bn0, an0);
    an1 = WMMA_BF16(a, bn1, an1);
  }

  const int hi = lane >> 4, nn = lane & 15;
  const int b = (m0 >> 3) + hi;  // 2 batches per 16-row tile
  const float* gib = gi + (size_t)b * D3;

  // Per-row (k-slot) index/weight are uniform across d; load once.
  int   slot[8];
  float wgt[8];
#pragma unroll
  for (int r = 0; r < 8; ++r) {
    const int m = m0 + r + (hi << 3);  // == b*8 + r
    slot[r] = tidx[m];
    wgt[r]  = tw[m];
  }

  v8f aR[2] = {ar0, ar1}, aZ[2] = {az0, az1}, aN[2] = {an0, an1};
#pragma unroll
  for (int c = 0; c < 2; ++c) {
    const int d = d0 + c * 16 + nn;
    const float gr = gib[d], gz = gib[Dd + d], gn = gib[2*Dd + d];
    const float brr = bhh[d], bzz = bhh[Dd + d], bnn = bhh[2*Dd + d];
    float u = 0.f;
#pragma unroll
    for (int r = 0; r < 8; ++r) {
      const float rg = 1.f / (1.f + expf(-(gr + aR[c][r] + brr)));
      const float zg = 1.f / (1.f + expf(-(gz + aZ[c][r] + bzz)));
      const float ng = tanhf(gn + rg * (aN[c][r] + bnn));
      const size_t soff = ((size_t)b * NS + slot[r]) * Dd + d;
      const float sold = S[soff];
      const float upd = (1.f - zg) * ng + zg * sold;
      Snew[soff] = upd;  // top_k indices are distinct per row -> no conflicts
      u += wgt[r] * upd;
    }
    Ub[(size_t)b * Dd + d] = (__bf16)u;
  }
}

// ---------------------------------------------------------------------------
// slot_mean over 128 slots + build bf16 ctx = [x, mean].  1 block per batch row.
// ---------------------------------------------------------------------------
__global__ __launch_bounds__(256) void build_ctx_kernel(
    const float* __restrict__ x, const float* __restrict__ S,
    __bf16* __restrict__ ctx) {
  const int b = blockIdx.x;
  const int t = threadIdx.x;  // 256 threads * float4 = 1024 cols
  const float4* Sp = (const float4*)(S + (size_t)b * NS * Dd);
  float4 acc = {0.f, 0.f, 0.f, 0.f};
#pragma unroll 4
  for (int s = 0; s < NS; ++s) {
    float4 v = Sp[s * 256 + t];
    acc.x += v.x; acc.y += v.y; acc.z += v.z; acc.w += v.w;
  }
  const float inv = 1.f / (float)NS;
  __bf16* cm = ctx + (size_t)b * D2;
  const float4 xv = ((const float4*)(x + (size_t)b * Dd))[t];
  const int d = t * 4;
  cm[d + 0] = (__bf16)xv.x;  cm[d + 1] = (__bf16)xv.y;
  cm[d + 2] = (__bf16)xv.z;  cm[d + 3] = (__bf16)xv.w;
  cm[Dd + d + 0] = (__bf16)(acc.x * inv);
  cm[Dd + d + 1] = (__bf16)(acc.y * inv);
  cm[Dd + d + 2] = (__bf16)(acc.z * inv);
  cm[Dd + d + 3] = (__bf16)(acc.w * inv);
}

// ---------------------------------------------------------------------------
// Top-8 of 128 + softmax, one wave32 per row (4 logits per lane).
// Tie-break = smaller index (matches jax.lax.top_k).
// ---------------------------------------------------------------------------
__global__ __launch_bounds__(256) void topk_softmax_kernel(
    const float* __restrict__ logits, int* __restrict__ idx_out,
    float* __restrict__ w_out) {
  const int row = blockIdx.x * 8 + (threadIdx.x >> 5);
  const int lane = threadIdx.x & 31;
  const float* lp = logits + (size_t)row * NS;
  float v[4]; int id[4];
#pragma unroll
  for (int j = 0; j < 4; ++j) { int c = lane + 32 * j; v[j] = lp[c]; id[j] = c; }
  float tv[TOPK]; int ti[TOPK];
  for (int t = 0; t < TOPK; ++t) {
    float best = -INFINITY; int bid = 1 << 30;
#pragma unroll
    for (int j = 0; j < 4; ++j)
      if (v[j] > best || (v[j] == best && id[j] < bid)) { best = v[j]; bid = id[j]; }
    for (int off = 16; off; off >>= 1) {
      float ov = __shfl_xor(best, off, 32);
      int   oi = __shfl_xor(bid,  off, 32);
      if (ov > best || (ov == best && oi < bid)) { best = ov; bid = oi; }
    }
    tv[t] = best; ti[t] = bid;
#pragma unroll
    for (int j = 0; j < 4; ++j) if (id[j] == bid) v[j] = -INFINITY;
  }
  if (lane < TOPK) {
    float s = 0.f;
#pragma unroll
    for (int t = 0; t < TOPK; ++t) s += expf(tv[t] - tv[0]);
    w_out[(size_t)row * TOPK + lane] = expf(tv[lane] - tv[0]) / s;
    idx_out[(size_t)row * TOPK + lane] = ti[lane];
  }
}

// ---------------------------------------------------------------------------
// Gather selected slot states -> bf16 rows [16384,1024]
// ---------------------------------------------------------------------------
__global__ __launch_bounds__(256) void gather_slots_kernel(
    const float* __restrict__ S, const int* __restrict__ tidx,
    __bf16* __restrict__ ss) {
  const int row = blockIdx.x;       // 16384
  const int b = row >> 3;
  const int slot = tidx[row];
  const float4* sp = (const float4*)(S + ((size_t)b * NS + slot) * Dd);
  __bf16* op = ss + (size_t)row * Dd;
  const int t = threadIdx.x;
  float4 v = sp[t];
  op[t * 4 + 0] = (__bf16)v.x; op[t * 4 + 1] = (__bf16)v.y;
  op[t * 4 + 2] = (__bf16)v.z; op[t * 4 + 3] = (__bf16)v.w;
}

// f32 -> bf16 (n divisible by 4)
__global__ __launch_bounds__(256) void cvt_bf16_kernel(
    const float* __restrict__ in, __bf16* __restrict__ out, int n) {
  const int i = (blockIdx.x * blockDim.x + threadIdx.x) * 4;
  if (i < n) {
    float4 v = *(const float4*)(in + i);
    out[i + 0] = (__bf16)v.x; out[i + 1] = (__bf16)v.y;
    out[i + 2] = (__bf16)v.z; out[i + 3] = (__bf16)v.w;
  }
}

// ---------------------------------------------------------------------------
extern "C" void kernel_launch(void* const* d_in, const int* in_sizes, int n_in,
                              void* d_out, int out_size, void* d_ws, size_t ws_size,
                              hipStream_t stream) {
  const float* x    = (const float*)d_in[0];
  const float* S    = (const float*)d_in[1];
  const float* w1   = (const float*)d_in[2];
  const float* b1   = (const float*)d_in[3];
  const float* w2   = (const float*)d_in[4];
  const float* b2   = (const float*)d_in[5];
  const float* wih  = (const float*)d_in[6];
  const float* whh  = (const float*)d_in[7];
  const float* bih  = (const float*)d_in[8];
  const float* bhh  = (const float*)d_in[9];
  const float* wval = (const float*)d_in[10];
  const float* bval = (const float*)d_in[11];
  const float* wout = (const float*)d_in[12];
  const float* bout = (const float*)d_in[13];
  const float* tau  = (const float*)d_in[14];

  float* out  = (float*)d_out;                 // [B,D]
  float* Snew = out + (size_t)Bsz * Dd;        // [B,128,D]

  char* ws = (char*)d_ws;
  size_t off = 0;
  auto alloc = [&](size_t bytes) -> void* {
    void* p = ws + off;
    off = (off + bytes + 255) & ~(size_t)255;
    return p;
  };
  __bf16* ctxb  = (__bf16*)alloc((size_t)Bsz * D2 * 2);
  __bf16* w1b   = (__bf16*)alloc((size_t)DH * D2 * 2);
  __bf16* w2b   = (__bf16*)alloc((size_t)NS * DH * 2);
  __bf16* hb    = (__bf16*)alloc((size_t)Bsz * DH * 2);
  float*  logit = (float*) alloc((size_t)Bsz * NS * 4);
  int*    tidx  = (int*)   alloc((size_t)BK * 4);
  float*  tw    = (float*) alloc((size_t)BK * 4);
  __bf16* wihb  = (__bf16*)alloc((size_t)D3 * Dd * 2);
  __bf16* whhb  = (__bf16*)alloc((size_t)D3 * Dd * 2);
  float*  gi    = (float*) alloc((size_t)Bsz * D3 * 4);
  __bf16* ssb   = (__bf16*)alloc((size_t)BK * Dd * 2);
  __bf16* Ub    = (__bf16*)alloc((size_t)Bsz * Dd * 2);
  __bf16* mixb  = (__bf16*)alloc((size_t)Bsz * Dd * 2);
  __bf16* wvalb = (__bf16*)alloc((size_t)Dd * Dd * 2);
  __bf16* woutb = (__bf16*)alloc((size_t)Dd * Dd * 2);
  (void)ws_size; (void)n_in; (void)in_sizes; (void)out_size;

  // S_new starts as a copy of S (updated rows scattered later)
  hipMemcpyAsync(Snew, S, (size_t)Bsz * NS * Dd * sizeof(float),
                 hipMemcpyDeviceToDevice, stream);

  auto cvt = [&](const float* src, __bf16* dst, size_t n) {
    cvt_bf16_kernel<<<dim3((unsigned)((n / 4 + 255) / 256)), 256, 0, stream>>>(src, dst, (int)n);
  };
  cvt(w1,   w1b,   (size_t)DH * D2);
  cvt(w2,   w2b,   (size_t)NS * DH);
  cvt(wih,  wihb,  (size_t)D3 * Dd);
  cvt(whh,  whhb,  (size_t)D3 * Dd);
  cvt(wval, wvalb, (size_t)Dd * Dd);
  cvt(wout, woutb, (size_t)Dd * Dd);

  // slot mean + bf16 ctx
  build_ctx_kernel<<<Bsz, 256, 0, stream>>>(x, S, ctxb);

  // h = gelu(ctx @ w1^T + b1) -> bf16 [2048,512]
  wmma_gemm_kernel<2><<<dim3(DH / 64, Bsz / 128), 256, 0, stream>>>(
      ctxb, D2, w1b, D2, b1, nullptr, nullptr, hb, DH, D2);

  // logits = (h @ w2^T + b2) / (|tau|+0.1) -> f32 [2048,128]
  wmma_gemm_kernel<0><<<dim3(NS / 64, Bsz / 128), 256, 0, stream>>>(
      hb, DH, w2b, DH, b2, tau, logit, nullptr, NS, DH);

  // top-8 + softmax
  topk_softmax_kernel<<<Bsz / 8, 256, 0, stream>>>(logit, tidx, tw);

  // gi = x @ wih^T + bih -> f32 [2048,3072]   (A = first 1024 cols of ctx)
  wmma_gemm_kernel<0><<<dim3(D3 / 64, Bsz / 128), 256, 0, stream>>>(
      ctxb, D2, wihb, Dd, bih, nullptr, gi, nullptr, D3, Dd);

  // gather selected states -> bf16 [16384,1024]
  gather_slots_kernel<<<BK, 256, 0, stream>>>(S, tidx, ssb);

  // fused 3-gate GRU GEMM + scatter + weighted readout state U
  gru_fused_kernel<<<dim3(Dd / 32, BK / 128), 256, 0, stream>>>(
      ssb, whhb, bhh, gi, S, tidx, tw, Snew, Ub);

  // mixed = U @ wval^T + bval -> bf16 [2048,1024]
  wmma_gemm_kernel<1><<<dim3(Dd / 64, Bsz / 128), 256, 0, stream>>>(
      Ub, Dd, wvalb, Dd, bval, nullptr, nullptr, mixb, Dd, Dd);

  // output = mixed @ wout^T + bout -> f32 into d_out
  wmma_gemm_kernel<0><<<dim3(Dd / 64, Bsz / 128), 256, 0, stream>>>(
      mixb, Dd, woutb, Dd, bout, nullptr, out, nullptr, Dd, Dd);
}
